// MultiHeadAttention_19224273617233
// MI455X (gfx1250) — matrix operand order repaired and checked by reading, not compiled, not run on hardware
//
#include <hip/hip_runtime.h>
#include <hip/hip_bf16.h>
#include <cstdint>

typedef __bf16 v16bf __attribute__((ext_vector_type(16)));
typedef __bf16 v8bf  __attribute__((ext_vector_type(8)));
typedef float  v8f   __attribute__((ext_vector_type(8)));

constexpr int kS1   = 2048;
constexpr int kS2   = 2048;
constexpr int kDin  = 1024;
constexpr int kDout = 1024;
constexpr int kH    = 16;
constexpr int kDK   = 64;
constexpr int kL    = 130;   // 2*RADIUS + 1 + 1 (segmented)
constexpr int kRad  = 64;

// ---------------------------------------------------------------------------
// One-shot fp32 -> bf16 conversion (removes all cvt work from GEMM hot loops)
// ---------------------------------------------------------------------------
__global__ __launch_bounds__(256)
void cvt_bf16_kernel(const float* __restrict__ src, __bf16* __restrict__ dst, int n)
{
    int i = blockIdx.x * blockDim.x + threadIdx.x;
    const int stride = gridDim.x * blockDim.x;
    for (; i < n; i += stride) dst[i] = (__bf16)src[i];
}

// ---------------------------------------------------------------------------
// Generic bf16 GEMM:  C[m][n] = sum_k A[m][k] * B[n][k]
//   mode 0: scatter C -> Kp[h][m][dk]  (bf16), with n = dk*16 + h
//   mode 1: scatter C -> Vt[h][dk][m]  (bf16, transposed over m)
//   mode 2: C -> outF[m][n]            (fp32)
// Block = 4 waves; block tile 64(M) x 64(N); K-step 32.
// B tile (64x32 bf16 = 4KB) is double-buffered in LDS and filled with
// GLOBAL_LOAD_ASYNC_TO_LDS_B128 (ASYNCcnt pipeline) shared by all 4 waves.
// ---------------------------------------------------------------------------
__global__ __launch_bounds__(128)
void gemm_bf16_kernel(const __bf16* __restrict__ A, const __bf16* __restrict__ B,
                      int Kdim, int mode,
                      __bf16* __restrict__ outKp, __bf16* __restrict__ outVt,
                      float* __restrict__ outF)
{
    // Bs[buf][n][k] : column n contiguous in k -> B-operand stripe loads
    __shared__ __align__(64) __bf16 Bs[2][64][32];

    const int lane = threadIdx.x & 31;
    const int wid  = threadIdx.x >> 5;
    const int m0   = blockIdx.x * 64 + wid * 16;
    const int n0   = blockIdx.y * 64;

    const int rl    = lane & 15;
    const int akoff = (lane < 16) ? 0 : 16;
    const int bkoff = (lane < 16) ? 0 : 16;

    // --- async B-tile staging setup: 256 chunks of 16B, 2 per thread ---
    const uint64_t Bbase = (uint64_t)(uintptr_t)(B + (size_t)n0 * Kdim);
    const int c0 = threadIdx.x;          // chunk ids
    const int c1 = threadIdx.x + 128;
    const int n_c0 = c0 >> 2, kk_c0 = (c0 & 3) * 8;
    const int n_c1 = c1 >> 2, kk_c1 = (c1 & 3) * 8;
    const uint32_t vb0 = (uint32_t)((n_c0 * Kdim + kk_c0) * 2);  // byte offs in B
    const uint32_t vb1 = (uint32_t)((n_c1 * Kdim + kk_c1) * 2);
    const uint32_t ldsBase0 = (uint32_t)(uintptr_t)(&Bs[0][0][0]);
    const uint32_t ldsBase1 = (uint32_t)(uintptr_t)(&Bs[1][0][0]);
    const uint32_t lo0 = (uint32_t)(c0 * 16);   // byte offset inside one buffer
    const uint32_t lo1 = (uint32_t)(c1 * 16);

    // issue stage 0 into buffer 0
    {
        uint32_t l0 = ldsBase0 + lo0, l1 = ldsBase0 + lo1;
        asm volatile("global_load_async_to_lds_b128 %0, %1, %2"
                     :: "v"(l0), "v"(vb0), "s"(Bbase) : "memory");
        asm volatile("global_load_async_to_lds_b128 %0, %1, %2"
                     :: "v"(l1), "v"(vb1), "s"(Bbase) : "memory");
    }

    v8f acc[4];
    v8f zero = {};
    #pragma unroll
    for (int nt = 0; nt < 4; ++nt) acc[nt] = zero;

    int it = 0;
    for (int kb = 0; kb < Kdim; kb += 32, ++it) {
        const bool has_next = (kb + 32) < Kdim;
        if (has_next) {
            // stage it+1 -> buffer (it+1)&1
            const uint32_t nb = ((it + 1) & 1) ? ldsBase1 : ldsBase0;
            const uint32_t ko = (uint32_t)((kb + 32) * 2);
            uint32_t l0 = nb + lo0, l1 = nb + lo1;
            uint32_t v0 = vb0 + ko,  v1 = vb1 + ko;
            asm volatile("global_load_async_to_lds_b128 %0, %1, %2"
                         :: "v"(l0), "v"(v0), "s"(Bbase) : "memory");
            asm volatile("global_load_async_to_lds_b128 %0, %1, %2"
                         :: "v"(l1), "v"(v1), "s"(Bbase) : "memory");
            asm volatile("s_wait_asynccnt 0x2" ::: "memory");  // current stage done
        } else {
            asm volatile("s_wait_asynccnt 0x0" ::: "memory");
        }
        __syncthreads();

        const int buf = it & 1;

        // A fragment (bf16, contiguous): lane<16 -> K=0..15; else K=16..31
        const __bf16* ap = A + (size_t)(m0 + rl) * Kdim + kb;
        v8bf alo = *(const v8bf*)(ap + akoff);
        v8bf ahi = *(const v8bf*)(ap + 8 + akoff);
        v16bf a;
        #pragma unroll
        for (int t = 0; t < 8; ++t) { a[t] = alo[t]; a[t + 8] = ahi[t]; }

        #pragma unroll
        for (int nt = 0; nt < 4; ++nt) {
            v16bf b = *(const v16bf*)(&Bs[buf][nt * 16 + rl][bkoff]);
            acc[nt] = __builtin_amdgcn_wmma_f32_16x16x32_bf16(
                false, a, false, b, (short)0, acc[nt], false, false);
        }
        __syncthreads();   // reads done before next iter's async overwrites buf
    }

    // C layout: VGPR r -> rows (r | r+8) across lane halves, N = lane&15
    const int mbase = m0 + ((lane < 16) ? 0 : 8);
    #pragma unroll
    for (int nt = 0; nt < 4; ++nt) {
        const int c  = n0 + nt * 16 + rl;
        const int h  = c & 15;
        const int dk = c >> 4;
        #pragma unroll
        for (int r = 0; r < 8; ++r) {
            const int m = mbase + r;
            const float v = acc[nt][r];
            if (mode == 0)      outKp[((size_t)h * kS2 + m) * kDK + dk] = (__bf16)v;
            else if (mode == 1) outVt[((size_t)h * kDK + dk) * kS2 + m] = (__bf16)v;
            else                outF[(size_t)m * kDout + c] = v;
        }
    }
}

// ---------------------------------------------------------------------------
// Flash attention with clipped relative bias + segment gating.
// Block = 4 waves; wave handles 16 query rows over all 2048 keys (32/iter).
// ---------------------------------------------------------------------------
__global__ __launch_bounds__(128)
void attn_kernel(const float* __restrict__ Q,
                 const __bf16* __restrict__ Kp,
                 const __bf16* __restrict__ Vt,
                 const long long* __restrict__ seg,
                 const unsigned char* __restrict__ pmask,
                 const float* __restrict__ rel,
                 __bf16* __restrict__ heads)
{
    __shared__ float  qrow[4][16][64];     // Qp rows for this wave
    __shared__ float  qe[4][16][132];      // bias table qe[i][l] (pad 130->132)
    __shared__ __bf16 pbuf[4][16][32];     // probability tile C->A relayout

    const int lane = threadIdx.x & 31;
    const int wid  = threadIdx.x >> 5;
    const int h    = blockIdx.y;
    const int i0   = blockIdx.x * 64 + wid * 16;

    // Stage Qp[h][i][d] = Q[i][d*16+h]
    for (int t = lane; t < 16 * 64; t += 32) {
        const int r = t >> 6, d = t & 63;
        qrow[wid][r][d] = Q[(size_t)(i0 + r) * kDout + d * kH + h];
    }
    __syncthreads();

    // qe[r][l] = dot(Qp row, rel_emb[h][l])
    for (int t = lane; t < 16 * kL; t += 32) {
        const int r = t / kL, l = t % kL;
        const float* rp = rel + ((size_t)h * kL + l) * kDK;
        float s = 0.f;
        #pragma unroll 8
        for (int d = 0; d < kDK; ++d) s += qrow[wid][r][d] * rp[d];
        qe[wid][r][l] = s;
    }
    __syncthreads();

    const int rl      = lane & 15;
    const int akoff   = (lane < 16) ? 0 : 16;
    const int bkoff   = (lane < 16) ? 0 : 16;
    const int halfoff = (lane < 16) ? 0 : 8;

    // Persistent Q A-fragments (d = 0..31 and 32..63)
    v16bf aq0, aq1;
    #pragma unroll
    for (int t = 0; t < 8; ++t) {
        aq0[t]     = (__bf16)qrow[wid][rl][akoff + t];
        aq0[t + 8] = (__bf16)qrow[wid][rl][8 + akoff + t];
        aq1[t]     = (__bf16)qrow[wid][rl][32 + akoff + t];
        aq1[t + 8] = (__bf16)qrow[wid][rl][40 + akoff + t];
    }

    float mrun[8], lrun[8];
    long long segi[8];
    v8f O[4];
    v8f zero = {};
    #pragma unroll
    for (int dt = 0; dt < 4; ++dt) O[dt] = zero;
    #pragma unroll
    for (int r = 0; r < 8; ++r) {
        mrun[r] = -3.0e38f;
        lrun[r] = 0.f;
        segi[r] = seg[i0 + halfoff + r];
    }

    const __bf16* __restrict__ Kph = Kp + (size_t)h * kS2 * kDK;
    const __bf16* __restrict__ Vth = Vt + (size_t)h * kDK * kS2;

    for (int j0 = 0; j0 < kS2; j0 += 32) {
        // ---- S = Q . K^T for 32 columns (two 16-wide C tiles) ----
        v8f s0 = zero, s1 = zero;
        {
            const __bf16* kc0 = Kph + (size_t)(j0 + rl) * kDK;
            const __bf16* kc1 = Kph + (size_t)(j0 + 16 + rl) * kDK;
            v16bf b;
            b  = *(const v16bf*)(kc0 + bkoff);
            s0 = __builtin_amdgcn_wmma_f32_16x16x32_bf16(false, aq0, false, b, (short)0, s0, false, false);
            b  = *(const v16bf*)(kc0 + 32 + bkoff);
            s0 = __builtin_amdgcn_wmma_f32_16x16x32_bf16(false, aq1, false, b, (short)0, s0, false, false);
            b  = *(const v16bf*)(kc1 + bkoff);
            s1 = __builtin_amdgcn_wmma_f32_16x16x32_bf16(false, aq0, false, b, (short)0, s1, false, false);
            b  = *(const v16bf*)(kc1 + 32 + bkoff);
            s1 = __builtin_amdgcn_wmma_f32_16x16x32_bf16(false, aq1, false, b, (short)0, s1, false, false);
        }

        // ---- bias + mask + online softmax (fp32) ----
        const long long sj0 = seg[j0 + rl];
        const long long sj1 = seg[j0 + 16 + rl];
        #pragma unroll
        for (int r = 0; r < 8; ++r) {
            const int mr = halfoff + r;
            const int i  = i0 + mr;
            const int ja = j0 + rl;
            const int jb = ja + 16;
            int ca = ja - i; ca = ca < -kRad ? -kRad : (ca > kRad ? kRad : ca); ca += kRad;
            int cb = jb - i; cb = cb < -kRad ? -kRad : (cb > kRad ? kRad : cb); cb += kRad;
            if (segi[r] != sj0) ca = 2 * kRad + 1;
            if (segi[r] != sj1) cb = 2 * kRad + 1;
            float sc0 = (s0[r] + qe[wid][mr][ca]) * 0.125f;  // 1/sqrt(64)
            float sc1 = (s1[r] + qe[wid][mr][cb]) * 0.125f;
            if (pmask[(size_t)i * kS2 + ja]) sc0 = -1.0e9f;
            if (pmask[(size_t)i * kS2 + jb]) sc1 = -1.0e9f;

            float tm = fmaxf(sc0, sc1);
            tm = fmaxf(tm, __shfl_xor(tm, 1));
            tm = fmaxf(tm, __shfl_xor(tm, 2));
            tm = fmaxf(tm, __shfl_xor(tm, 4));
            tm = fmaxf(tm, __shfl_xor(tm, 8));
            const float mn   = fmaxf(mrun[r], tm);
            const float corr = __expf(mrun[r] - mn);
            mrun[r] = mn;
            const float p0 = __expf(sc0 - mn);
            const float p1 = __expf(sc1 - mn);
            float rs = p0 + p1;
            rs += __shfl_xor(rs, 1);
            rs += __shfl_xor(rs, 2);
            rs += __shfl_xor(rs, 4);
            rs += __shfl_xor(rs, 8);
            lrun[r] = lrun[r] * corr + rs;
            #pragma unroll
            for (int dt = 0; dt < 4; ++dt) O[dt][r] *= corr;
            pbuf[wid][mr][rl]      = (__bf16)p0;
            pbuf[wid][mr][16 + rl] = (__bf16)p1;
        }
        __syncthreads();

        // ---- P as A-operand, O += P . V ----
        v16bf pa;
        #pragma unroll
        for (int t = 0; t < 8; ++t) {
            pa[t]     = pbuf[wid][rl][akoff + t];
            pa[t + 8] = pbuf[wid][rl][8 + akoff + t];
        }
        #pragma unroll
        for (int dt = 0; dt < 4; ++dt) {
            const __bf16* vp = Vth + (size_t)(dt * 16 + rl) * kS2 + j0 + bkoff;
            v16bf b = *(const v16bf*)vp;
            O[dt] = __builtin_amdgcn_wmma_f32_16x16x32_bf16(
                false, pa, false, b, (short)0, O[dt], false, false);
        }
        __syncthreads();
    }

    // heads[i][dk*16 + h] = O / l   (bf16 for the final bf16 GEMM)
    #pragma unroll
    for (int dt = 0; dt < 4; ++dt) {
        const int dk = dt * 16 + rl;
        #pragma unroll
        for (int r = 0; r < 8; ++r) {
            const int i = i0 + halfoff + r;
            heads[(size_t)i * kDout + dk * kH + h] = (__bf16)(O[dt][r] / lrun[r]);
        }
    }
}

// ---------------------------------------------------------------------------
extern "C" void kernel_launch(void* const* d_in, const int* in_sizes, int n_in,
                              void* d_out, int out_size, void* d_ws, size_t ws_size,
                              hipStream_t stream)
{
    (void)in_sizes; (void)n_in; (void)out_size; (void)ws_size;
    const float*         Q   = (const float*)d_in[0];
    const float*         K   = (const float*)d_in[1];
    const float*         V   = (const float*)d_in[2];
    const long long*     seg = (const long long*)d_in[3];
    const unsigned char* pm  = (const unsigned char*)d_in[4];
    const float*         Wk  = (const float*)d_in[5];
    const float*         Wv  = (const float*)d_in[6];
    const float*         Wc  = (const float*)d_in[7];
    const float*         rel = (const float*)d_in[8];
    float*               out = (float*)d_out;

    // Workspace (bf16 elements):
    //  Kbf 2M | Vbf 2M | Wkbf 1M | Wvbf 1M | Wcbf 1M | Kp 2M | Vt 2M | heads 2M
    __bf16* Kbf   = (__bf16*)d_ws;
    __bf16* Vbf   = Kbf  + (size_t)kS2 * kDin;
    __bf16* Wkbf  = Vbf  + (size_t)kS2 * kDin;
    __bf16* Wvbf  = Wkbf + (size_t)kDout * kDin;
    __bf16* Wcbf  = Wvbf + (size_t)kDout * kDin;
    __bf16* Kp    = Wcbf + (size_t)kDout * kDout;
    __bf16* Vt    = Kp   + (size_t)kH * kS2 * kDK;
    __bf16* heads = Vt   + (size_t)kH * kDK * kS2;

    cvt_bf16_kernel<<<1024, 256, 0, stream>>>(K,  Kbf,  kS2 * kDin);
    cvt_bf16_kernel<<<1024, 256, 0, stream>>>(V,  Vbf,  kS2 * kDin);
    cvt_bf16_kernel<<<1024, 256, 0, stream>>>(Wk, Wkbf, kDout * kDin);
    cvt_bf16_kernel<<<1024, 256, 0, stream>>>(Wv, Wvbf, kDout * kDin);
    cvt_bf16_kernel<<<1024, 256, 0, stream>>>(Wc, Wcbf, kDout * kDout);

    gemm_bf16_kernel<<<dim3(kS2 / 64, kDout / 64), 128, 0, stream>>>(
        Kbf, Wkbf, kDin, 0, Kp, Vt, nullptr);
    gemm_bf16_kernel<<<dim3(kS2 / 64, kDout / 64), 128, 0, stream>>>(
        Vbf, Wvbf, kDin, 1, Kp, Vt, nullptr);

    attn_kernel<<<dim3(kS1 / 64, kH), 128, 0, stream>>>(Q, Kp, Vt, seg, pm, rel, heads);

    gemm_bf16_kernel<<<dim3(kS1 / 64, kDout / 64), 128, 0, stream>>>(
        heads, Wcbf, kDout, 2, nullptr, nullptr, out);
}
